// NMILArchitecture_41059887350324
// MI455X (gfx1250) — matrix operand order repaired and checked by reading, not compiled, not run on hardware
//
#include <hip/hip_runtime.h>
#include <hip/hip_bf16.h>
#include <math.h>

// ---------------------------------------------------------------------------
// Problem constants (from the reference)
// ---------------------------------------------------------------------------
#define N_INST 131072
#define L_DIM  512
#define D_DIM  128
#define R_REG  64
#define N2_DIM 256
#define N3_DIM 128
#define NC_DIM 2

typedef __attribute__((ext_vector_type(16))) __bf16 v16bf;
typedef __attribute__((ext_vector_type(8)))  __bf16 v8bf;
typedef __attribute__((ext_vector_type(8)))  float  v8f;

// Monotone float<->uint encoding so segment-max can use integer atomicMax
// (order-independent => deterministic).
__device__ __forceinline__ unsigned enc_f32(float x) {
    unsigned u = __float_as_uint(x);
    return (u & 0x80000000u) ? ~u : (u | 0x80000000u);
}
__device__ __forceinline__ float dec_f32(unsigned u) {
    return (u & 0x80000000u) ? __uint_as_float(u & 0x7fffffffu)
                             : __uint_as_float(~u);
}
#define ENC_NEG_MAX 0x00800000u // enc(-FLT_MAX)

__device__ __forceinline__ float sigmoidf_(float x) {
    return 1.0f / (1.0f + __expf(-x));
}

#define SHUF16(lo, hi) __builtin_shufflevector(lo, hi, \
    0, 1, 2, 3, 4, 5, 6, 7, 8, 9, 10, 11, 12, 13, 14, 15)

// ---------------------------------------------------------------------------
// Kernel 0: workspace init (workspace is poisoned by the harness)
// ---------------------------------------------------------------------------
__global__ void init_ws(unsigned* __restrict__ m_enc,
                        float* __restrict__ z,
                        unsigned* __restrict__ counts,
                        float* __restrict__ region_emb) {
    const int tid = threadIdx.x;
    if (tid < R_REG) {
        m_enc[tid]  = ENC_NEG_MAX;
        z[tid]      = 0.0f;
        counts[tid] = 0u;
    }
    for (int e = tid; e < R_REG * L_DIM; e += blockDim.x) region_emb[e] = 0.0f;
}

// ---------------------------------------------------------------------------
// Kernel 1/4: gated-attention MIL logits via bf16 WMMA.
//   w[i] = sum_d tanh(X@av_w + av_b)_d * sigmoid(X@au_w + au_b)_d * aw_w[d] + aw_b
//
// Gate-split layout: waves 0-3 run the tanh GEMM (av_w) for M-tiles 0-3,
// waves 4-7 run the sigmoid GEMM (au_w) for the SAME tiles. Per-wave
// accumulators = 64 VGPRs, so all 8 B fragments stay live: 16 ds_load_b128
// in flight per s_wait_dscnt, 8 back-to-back v_wmma. Double-buffered LDS
// staging with register prefetch hides HBM/L2 latency under the WMMAs.
// Gates are combined through a conflict-free (stride-65) LDS exchange that
// overlays the dead staging buffers.
// ---------------------------------------------------------------------------
#define LDP 40  // padded LDS row stride (elements): 80B => 16B aligned, 20-bank shift
#define RBB 64  // rows per block (4 M-tiles, each owned by a wave pair)

#define SA_ELEMS (RBB * LDP)        // 2560 bf16 per A buffer (5120 B)
#define SB_ELEMS (2 * 128 * LDP)    // 10240 bf16 per B buffer (20480 B)
#define SMEM_BYTES (2 * SA_ELEMS * 2 + 2 * SB_ELEMS * 2) // 51200; exchange uses 33280

__launch_bounds__(256)
__global__ void mil_logits_wmma(const float* __restrict__ X,
                                const float* __restrict__ av_w,
                                const float* __restrict__ av_b,
                                const float* __restrict__ au_w,
                                const float* __restrict__ au_b,
                                const float* __restrict__ aw_w,
                                const float* __restrict__ aw_b,
                                float* __restrict__ wout,
                                const int* __restrict__ seg,
                                unsigned* __restrict__ m_enc) {
    constexpr int NT  = 256;
    constexpr int APT = (RBB * 32) / NT;      // 8 A elements per thread
    constexpr int BPT = (2 * 32 * 128) / NT;  // 32 B elements per thread

    __shared__ __align__(16) char smem[SMEM_BYTES];
    // buffer addresses computed arithmetically (no pointer-array initializer,
    // which would require an addrspacecast in a static initializer)
    __bf16* sA = (__bf16*)smem;                           // 2 x SA_ELEMS
    __bf16* sB = (__bf16*)(smem + 2 * SA_ELEMS * 2);      // 2 x SB_ELEMS

    const int tid  = threadIdx.x;
    const int lane = tid & 31;
    const int wv   = tid >> 5;
    const int mt   = wv & 3;    // M-tile owned by this wave
    const int mat  = wv >> 2;   // 0: tanh GEMM (av_w), 1: sigmoid GEMM (au_w)
    const int row0 = blockIdx.x * RBB;
    const int half = lane >> 4; // lane group (0: lanes 0-15)
    const int nl   = lane & 15;

    v8f zero = {};
    v8f acc[8];
#pragma unroll
    for (int t = 0; t < 8; ++t) acc[t] = zero;

    float ta[APT], tb[BPT];

    auto global_fetch = [&](int kc) {
#pragma unroll
        for (int j = 0; j < APT; ++j) {
            const int e = tid + j * NT;
            const int r = e >> 5, c = e & 31;
            ta[j] = X[(row0 + r) * L_DIM + kc * 32 + c];   // coalesced
        }
#pragma unroll
        for (int j = 0; j < BPT; ++j) {
            const int e = tid + j * NT;
            const int m = e >> 12;
            const int k = (e >> 7) & 31;
            const int n = e & 127;
            const float* Wm = m ? au_w : av_w;
            tb[j] = Wm[(kc * 32 + k) * D_DIM + n];         // coalesced (L2)
        }
    };
    auto lds_store = [&](int buf) {
        __bf16* dA = sA + buf * SA_ELEMS;
        __bf16* dB = sB + buf * SB_ELEMS;
#pragma unroll
        for (int j = 0; j < APT; ++j) {
            const int e = tid + j * NT;
            const int r = e >> 5, c = e & 31;
            dA[r * LDP + c] = (__bf16)ta[j];
        }
#pragma unroll
        for (int j = 0; j < BPT; ++j) {
            const int e = tid + j * NT;
            const int m = e >> 12;
            const int k = (e >> 7) & 31;
            const int n = e & 127;
            dB[(m * 128 + n) * LDP + k] = (__bf16)tb[j];
        }
    };

    // prologue: stage chunk 0
    global_fetch(0);
    lds_store(0);
    __syncthreads();

    for (int kc = 0; kc < L_DIM / 32; ++kc) {
        const int cur = kc & 1;
        if (kc + 1 < L_DIM / 32) global_fetch(kc + 1);  // hide latency under WMMAs

        // ---- A fragment (ISA 16-bit A 16x32 layout):
        // lanes 0-15: K=0..7 (v0-3), 16..23 (v4-7); lanes 16-31: 8..15, 24..31
        const __bf16* pA = sA + cur * SA_ELEMS + (mt * 16 + nl) * LDP;
        v8bf alo = *(const v8bf*)(pA + half * 8);
        v8bf ahi = *(const v8bf*)(pA + 16 + half * 8);
        v16bf afrag = SHUF16(alo, ahi);

        // ---- all 8 B fragments live => loads batch under one s_wait_dscnt
        // B fragment (same K order as A): lanes 0-15 hold K=0..7, 16..23,
        // lanes 16-31 hold K=8..15, 24..31.
        const __bf16* pB = sB + cur * SB_ELEMS + (mat * 128 + nl) * LDP + half * 8;
        v16bf bfr[8];
#pragma unroll
        for (int nt = 0; nt < 8; ++nt) {
            const __bf16* p = pB + nt * 16 * LDP;
            bfr[nt] = SHUF16(*(const v8bf*)p, *(const v8bf*)(p + 16));
        }
#pragma unroll
        for (int nt = 0; nt < 8; ++nt)
            acc[nt] = __builtin_amdgcn_wmma_f32_16x16x32_bf16(
                false, afrag, false, bfr[nt], (short)0, acc[nt], false, false);

        if (kc + 1 < L_DIM / 32) lds_store(cur ^ 1);  // disjoint buffer: no race
        __syncthreads();
    }

    // ---- epilogue. C layout: n = lane%16 (+16*nt), m = vgpr + 8*(lane/16).
    // Paired waves hold matching elements at identical (lane, vgpr) indices,
    // so the exchange is a direct store/load at the same LDS slot.
    float* ex = (float*)smem;  // overlays dead staging buffers (33280 B used)
    const int exbase = (mt * 32 + lane) * 65;  // stride 65 => conflict-free

    if (mat == 1) {  // sigmoid wave: publish gated values
#pragma unroll
        for (int nt = 0; nt < 8; ++nt) {
            const float bu = au_b[nt * 16 + nl];
#pragma unroll
            for (int r = 0; r < 8; ++r)
                ex[exbase + nt * 8 + r] = sigmoidf_(acc[nt][r] + bu);
        }
    }
    __syncthreads();

    if (mat == 0) {  // tanh wave: combine, reduce over D, emit logits
        const float awb0 = aw_b[0];
        float s[8];
#pragma unroll
        for (int r = 0; r < 8; ++r) s[r] = 0.0f;
#pragma unroll
        for (int nt = 0; nt < 8; ++nt) {
            const int n  = nt * 16 + nl;
            const float bv = av_b[n];
            const float ww = aw_w[n];
#pragma unroll
            for (int r = 0; r < 8; ++r) {
                const float g = ex[exbase + nt * 8 + r];
                s[r] += tanhf(acc[nt][r] + bv) * g * ww;
            }
        }
#pragma unroll
        for (int r = 0; r < 8; ++r) {
            float v = s[r];
            v += __shfl_xor(v, 1, 32);
            v += __shfl_xor(v, 2, 32);
            v += __shfl_xor(v, 4, 32);
            v += __shfl_xor(v, 8, 32);  // stays within each 16-lane group
            s[r] = v;
        }
        if (nl == 0) {  // lanes 0 and 16 hold rows m=0..7 and m=8..15 resp.
            const int rowbase = row0 + mt * 16 + half * 8;
#pragma unroll
            for (int r = 0; r < 8; ++r) {
                const int row = rowbase + r;
                const float wval = s[r] + awb0;
                wout[row] = wval;
                if (m_enc != nullptr) {
                    const int reg = seg[row];
                    atomicMax(&m_enc[reg], enc_f32(wval));
                }
            }
        }
    }
}

// ---------------------------------------------------------------------------
// Kernel 2: e = exp(w - m[seg]); z[r] += e; counts[r] += 1  (LDS partials)
// ---------------------------------------------------------------------------
__launch_bounds__(256)
__global__ void softmax_stats(const float* __restrict__ w,
                              const int* __restrict__ seg,
                              const unsigned* __restrict__ m_enc,
                              float* __restrict__ ebuf,
                              float* __restrict__ z,
                              unsigned* __restrict__ counts) {
    __shared__ float    zs[R_REG];
    __shared__ unsigned cs[R_REG];
    const int tid = threadIdx.x;
    if (tid < R_REG) { zs[tid] = 0.0f; cs[tid] = 0u; }
    __syncthreads();

    const int i = blockIdx.x * blockDim.x + tid;
    const int r = seg[i];
    const float m  = dec_f32(m_enc[r]);
    const float ev = __expf(w[i] - m);
    ebuf[i] = ev;
    atomicAdd(&zs[r], ev);     // ds_add_f32
    atomicAdd(&cs[r], 1u);     // ds_add_u32
    __syncthreads();

    if (tid < R_REG) {
        atomicAdd(&z[tid], zs[tid]);
        atomicAdd(&counts[tid], cs[tid]);
    }
}

// ---------------------------------------------------------------------------
// Kernel 3: patch outputs + attention-pooled region embeddings.
// Single pass over features (268 MB -> one HBM read, ~11.5us at 23.3 TB/s).
// Serial-owner scheme: thread t owns columns t and t+256 of the 128 KB LDS
// accumulator (MI455X WGP has 320 KB) => no atomics / races in the hot loop.
// ---------------------------------------------------------------------------
#define POOL_IPB 1024  // instances per block (N_INST / 128 blocks)

__launch_bounds__(256)
__global__ void pool_regions(const float* __restrict__ X,
                             const float* __restrict__ wbuf,
                             const float* __restrict__ ebuf,
                             const int* __restrict__ seg,
                             const float* __restrict__ z,
                             const unsigned* __restrict__ counts,
                             float* __restrict__ patch_out,
                             float* __restrict__ region_emb) {
    extern __shared__ float racc[];  // R_REG * L_DIM floats = 128 KB
    const int tid = threadIdx.x;
    const int i0  = blockIdx.x * POOL_IPB;

    // patch_classification: raw logit for singleton regions, softmax otherwise
#pragma unroll
    for (int q = 0; q < POOL_IPB / 256; ++q) {
        const int i = i0 + q * 256 + tid;
        const int r = seg[i];
        const float sm = ebuf[i] / z[r];
        patch_out[i] = (counts[r] == 1u) ? wbuf[i] : sm;
    }

    for (int e = tid; e < R_REG * L_DIM; e += 256) racc[e] = 0.0f;
    __syncthreads();

    for (int q = 0; q < POOL_IPB; ++q) {
        const int i = i0 + q;            // uniform across the block
        const int r = seg[i];
        const float sm = ebuf[i] / z[r];
        // thread t owns columns t and t+256 of every region => race-free
        racc[r * L_DIM + tid]       += sm * X[i * L_DIM + tid];
        racc[r * L_DIM + tid + 256] += sm * X[i * L_DIM + tid + 256];
    }
    __syncthreads();

    for (int e = tid; e < R_REG * L_DIM; e += 256)
        atomicAdd(&region_emb[e], racc[e]);
}

// ---------------------------------------------------------------------------
// Kernel 5: region softmax -> pooled embedding -> classifier MLP
// ---------------------------------------------------------------------------
__launch_bounds__(256)
__global__ void finalize(const float* __restrict__ region_emb,
                         const float* __restrict__ w_reg,
                         const float* __restrict__ c1_w, const float* __restrict__ c1_b,
                         const float* __restrict__ c2_w, const float* __restrict__ c2_b,
                         const float* __restrict__ c3_w, const float* __restrict__ c3_b,
                         float* __restrict__ out) {
    __shared__ float smr[R_REG];
    __shared__ float emb[L_DIM];
    __shared__ float h1[N2_DIM];
    __shared__ float h2[N3_DIM];
    const int tid = threadIdx.x;

    // softmax over the 64 region logits (redundant per-thread scan is cheap)
    float mx = -3.4e38f;
    for (int r = 0; r < R_REG; ++r) mx = fmaxf(mx, w_reg[r]);
    float zz = 0.0f;
    for (int r = 0; r < R_REG; ++r) zz += __expf(w_reg[r] - mx);
    if (tid < R_REG) smr[tid] = __expf(w_reg[tid] - mx) / zz;
    __syncthreads();

    // embedding[l] = sum_r region_emb[r][l] * smr[r]
    for (int l = tid; l < L_DIM; l += 256) {
        float a = 0.0f;
        for (int r = 0; r < R_REG; ++r) a += region_emb[r * L_DIM + l] * smr[r];
        emb[l] = a;
    }
    __syncthreads();

    {   // h1 = relu(emb @ c1_w + c1_b), one output per thread
        float a = c1_b[tid];
        for (int k = 0; k < L_DIM; ++k) a += emb[k] * c1_w[k * N2_DIM + tid];
        h1[tid] = fmaxf(a, 0.0f);
    }
    __syncthreads();
    if (tid < N3_DIM) {
        float a = c2_b[tid];
        for (int k = 0; k < N2_DIM; ++k) a += h1[k] * c2_w[k * N3_DIM + tid];
        h2[tid] = fmaxf(a, 0.0f);
    }
    __syncthreads();
    if (tid < NC_DIM) {
        float a = c3_b[tid];
        for (int k = 0; k < N3_DIM; ++k) a += h2[k] * c3_w[k * NC_DIM + tid];
        out[tid] = a;
    }
}

// ---------------------------------------------------------------------------
// Host launcher
// ---------------------------------------------------------------------------
extern "C" void kernel_launch(void* const* d_in, const int* in_sizes, int n_in,
                              void* d_out, int out_size, void* d_ws, size_t ws_size,
                              hipStream_t stream) {
    const float* features    = (const float*)d_in[0];
    const int*   region_info = (const int*)  d_in[1];
    // d_in[2] = num_regions (scalar, fixed at 64 here)
    const float* av_w = (const float*)d_in[3];
    const float* av_b = (const float*)d_in[4];
    const float* au_w = (const float*)d_in[5];
    const float* au_b = (const float*)d_in[6];
    const float* aw_w = (const float*)d_in[7];
    const float* aw_b = (const float*)d_in[8];
    const float* c1_w = (const float*)d_in[9];
    const float* c1_b = (const float*)d_in[10];
    const float* c2_w = (const float*)d_in[11];
    const float* c2_b = (const float*)d_in[12];
    const float* c3_w = (const float*)d_in[13];
    const float* c3_b = (const float*)d_in[14];

    // workspace layout (bytes)
    char* ws = (char*)d_ws;
    float*    w_buf      = (float*)   (ws);                          // N
    float*    e_buf      = (float*)   (ws + 524288);                 // N
    unsigned* m_enc      = (unsigned*)(ws + 1048576);                // R
    float*    z_buf      = (float*)   (ws + 1048576 + 256);          // R
    unsigned* counts     = (unsigned*)(ws + 1048576 + 512);          // R
    float*    region_emb = (float*)   (ws + 1048576 + 1024);         // R*L
    float*    w_reg      = (float*)   (ws + 1048576 + 1024 + 131072);// R

    float* out   = (float*)d_out;  // [0..1] global_classification
    float* patch = out + NC_DIM;   // [2..2+N) patch_classification

    // allow the 128 KB single-pass LDS accumulator (MI455X: 320 KB per WGP)
    (void)hipFuncSetAttribute((const void*)pool_regions,
                              hipFuncAttributeMaxDynamicSharedMemorySize,
                              R_REG * L_DIM * (int)sizeof(float));

    init_ws<<<1, 256, 0, stream>>>(m_enc, z_buf, counts, region_emb);

    // level-1 MIL logits: 2048 blocks x 8 waves (4 wave pairs), 64 rows/block
    mil_logits_wmma<<<N_INST / RBB, 256, 0, stream>>>(
        features, av_w, av_b, au_w, au_b, aw_w, aw_b,
        w_buf, region_info, m_enc);

    softmax_stats<<<N_INST / 256, 256, 0, stream>>>(
        w_buf, region_info, m_enc, e_buf, z_buf, counts);

    pool_regions<<<N_INST / POOL_IPB, 256, R_REG * L_DIM * sizeof(float), stream>>>(
        features, w_buf, e_buf, region_info, z_buf, counts,
        patch, region_emb);

    // level-2 MIL logits over the 64 region embeddings (one block, 64 rows)
    mil_logits_wmma<<<1, 256, 0, stream>>>(
        region_emb, av_w, av_b, au_w, au_b, aw_w, aw_b,
        w_reg, nullptr, nullptr);

    finalize<<<1, 256, 0, stream>>>(
        region_emb, w_reg, c1_w, c1_b, c2_w, c2_b, c3_w, c3_b, out);
}
